// Partition_enhanced_GIN_31482110280432
// MI455X (gfx1250) — matrix operand order repaired and checked by reading, not compiled, not run on hardware
//
#include <hip/hip_runtime.h>

#define NN 50000
#define NE 800000
#define NG 64
#define NL 3
#define NC 4
#define H  64
#define BN_EPS 1e-5f

typedef __attribute__((ext_vector_type(16))) _Float16 v16h;
typedef __attribute__((ext_vector_type(8)))  float    v8f;

// counters layout (ints): [0..3] node_cnt, [4..7] edge_cnt, [8..11] node_off,
// [12..15] edge_off, [16..19] node_cur, [20..23] edge_cur
// stats layout (floats): [0..63] sum, [64..127] sumsq, [128..191] scale, [192..255] shift

__global__ void k_init(const float* __restrict__ x, float* __restrict__ x_work,
                       float* __restrict__ pooled, int* __restrict__ cnts) {
  int i = blockIdx.x * blockDim.x + threadIdx.x;
  if (i < NN * H) x_work[i] = x[i];
  if (i < NG * NL * H) pooled[i] = 0.f;
  if (i < 32) cnts[i] = 0;
}

__global__ void k_bucket_count(const int* __restrict__ labels,
                               const long long* __restrict__ ei, int* __restrict__ cnts) {
  int i = blockIdx.x * blockDim.x + threadIdx.x;
  if (i < NN) atomicAdd(&cnts[labels[i]], 1);
  int e = i - NN;
  if (e >= 0 && e < NE) {
    int d = (int)ei[NE + e];
    atomicAdd(&cnts[4 + labels[d]], 1);
  }
}

__global__ void k_bucket_offsets(int* __restrict__ cnts) {
  if (threadIdx.x == 0 && blockIdx.x == 0) {
    int no = 0, eo = 0;
    for (int c = 0; c < NC; c++) {
      cnts[8 + c]  = no; cnts[16 + c] = no; no += cnts[c];
      cnts[12 + c] = eo; cnts[20 + c] = eo; eo += cnts[4 + c];
    }
  }
}

__global__ void k_bucket_fill(const int* __restrict__ labels,
                              const long long* __restrict__ ei, int* __restrict__ cnts,
                              int* __restrict__ node_list, int* __restrict__ edge_list) {
  int i = blockIdx.x * blockDim.x + threadIdx.x;
  if (i < NN) {
    int c = labels[i];
    int p = atomicAdd(&cnts[16 + c], 1);
    node_list[p] = i;
  }
  int e = i - NN;
  if (e >= 0 && e < NE) {
    int c = labels[(int)ei[NE + e]];
    int p = atomicAdd(&cnts[20 + c], 1);
    edge_list[p] = e;
  }
}

// zero agg rows for cluster c (only masked rows receive edge contributions) + zero BN stats
__global__ void k_zero_agg(int c, const int* __restrict__ cnts, const int* __restrict__ node_list,
                           float* __restrict__ agg, float* __restrict__ stats) {
  int cnt = cnts[c], off = cnts[8 + c];
  int total = cnt * H;
  for (int i = blockIdx.x * blockDim.x + threadIdx.x; i < total; i += gridDim.x * blockDim.x) {
    int node = node_list[off + (i >> 6)];
    agg[node * H + (i & 63)] = 0.f;
  }
  if (blockIdx.x == 0 && threadIdx.x < 128) stats[threadIdx.x] = 0.f;
}

// GIN aggregation restricted to edges whose dst is in cluster c.
// 16 threads per edge; float4 gathers (global_load_b128), 4 f32 atomics per thread.
__global__ void k_aggregate(int c, const int* __restrict__ cnts, const int* __restrict__ edge_list,
                            const long long* __restrict__ ei, const float* __restrict__ x_work,
                            float* __restrict__ agg) {
  int ecnt = cnts[4 + c], eoff = cnts[12 + c];
  long long total = (long long)ecnt * 16;
  long long stride = (long long)gridDim.x * blockDim.x;
  for (long long i = blockIdx.x * (long long)blockDim.x + threadIdx.x; i < total; i += stride) {
    int j = (int)(i >> 4);
    int q = (int)(i & 15) * 4;
    int e = edge_list[eoff + j];
    int s = (int)ei[e];
    int d = (int)ei[NE + e];
    const float4 v = *(const float4*)(x_work + s * H + q);
    float* ap = agg + d * H + q;
    atomicAdd(ap + 0, v.x);
    atomicAdd(ap + 1, v.y);
    atomicAdd(ap + 2, v.z);
    atomicAdd(ap + 3, v.w);
  }
}

// h_mid = (agg + x) @ W1[t,c] + b1[t,c]  over compact cluster rows; accumulate BN stats.
__global__ __launch_bounds__(256)
void k_gemm1(int t, int c, const int* __restrict__ cnts, const int* __restrict__ node_list,
             const float* __restrict__ agg, const float* __restrict__ x_work,
             const float* __restrict__ W1, const float* __restrict__ b1,
             float* __restrict__ h_mid, float* __restrict__ stats) {
  __shared__ _Float16 sA[128 * 72];       // A panel (padded stride 72 to dodge bank conflicts)
  __shared__ _Float16 sW[64 * 72];        // W transposed: sW[n*72+k] = W[k][n]
  int cnt = cnts[c], coff = cnts[8 + c];
  int rowBase = blockIdx.x * 128;
  if (rowBase >= cnt) return;
  bool fullTile = (rowBase + 128 <= cnt);

  const float* W = W1 + (t * NC + c) * H * H;
  for (int idx = threadIdx.x; idx < H * H; idx += 256) {
    int k = idx >> 6, n = idx & 63;
    sW[n * 72 + k] = (_Float16)W[idx];
  }
  {
    int r  = threadIdx.x >> 1;
    int k0 = (threadIdx.x & 1) * 32;
    int grow = rowBase + r;
    _Float16* dstp = &sA[r * 72 + k0];
    if (grow < cnt) {
      int node = node_list[coff + grow];
      const float* pa = agg + node * H + k0;
      const float* px = x_work + node * H + k0;
      #pragma unroll
      for (int i = 0; i < 32; i++) dstp[i] = (_Float16)(pa[i] + px[i]);
    } else {
      #pragma unroll
      for (int i = 0; i < 32; i++) dstp[i] = (_Float16)0.f;
    }
  }
  __syncthreads();

  int lane = threadIdx.x & 31;
  int wave = threadIdx.x >> 5;
  int m = lane & 15, half = lane >> 4;
  int r0 = wave * 16;
  v8f acc[4] = {};
  #pragma unroll
  for (int kc = 0; kc < 2; kc++) {
    v16h a;
    const _Float16* arow = &sA[(r0 + m) * 72 + kc * 32];
    #pragma unroll
    for (int i = 0; i < 16; i++) a[i] = arow[half * 16 + i];
    #pragma unroll
    for (int nt = 0; nt < 4; nt++) {
      v16h b;
      const _Float16* bcol = &sW[(nt * 16 + m) * 72 + kc * 32 + half * 16];
      #pragma unroll
      for (int i = 0; i < 16; i++) b[i] = bcol[i];
      acc[nt] = __builtin_amdgcn_wmma_f32_16x16x32_f16(false, a, false, b, (short)0,
                                                       acc[nt], false, false);
    }
  }
  const float* bias = b1 + (t * NC + c) * H;
  if (fullTile) {
    // uniform fast path: no per-row exec juggling
    #pragma unroll
    for (int nt = 0; nt < 4; nt++) {
      int ncol = nt * 16 + m;
      float bv = bias[ncol];
      float csum = 0.f, csq = 0.f;
      #pragma unroll
      for (int v = 0; v < 8; v++) {
        int grow = rowBase + r0 + half * 8 + v;
        float val = acc[nt][v] + bv;
        h_mid[grow * H + ncol] = val;
        csum += val; csq += val * val;
      }
      csum += __shfl_xor(csum, 16, 32);
      csq  += __shfl_xor(csq, 16, 32);
      if (half == 0) {
        atomicAdd(&stats[ncol], csum);
        atomicAdd(&stats[64 + ncol], csq);
      }
    }
  } else {
    #pragma unroll
    for (int nt = 0; nt < 4; nt++) {
      int ncol = nt * 16 + m;
      float bv = bias[ncol];
      float csum = 0.f, csq = 0.f;
      #pragma unroll
      for (int v = 0; v < 8; v++) {
        int grow = rowBase + r0 + half * 8 + v;
        if (grow < cnt) {
          float val = acc[nt][v] + bv;
          h_mid[grow * H + ncol] = val;
          csum += val; csq += val * val;
        }
      }
      csum += __shfl_xor(csum, 16, 32);
      csq  += __shfl_xor(csq, 16, 32);
      if (half == 0) {
        atomicAdd(&stats[ncol], csum);
        atomicAdd(&stats[64 + ncol], csq);
      }
    }
  }
}

__global__ void k_bn_finalize(int t, int c, const int* __restrict__ cnts,
                              const float* __restrict__ g1, const float* __restrict__ be1,
                              float* __restrict__ stats) {
  int f = threadIdx.x;
  if (f < H) {
    float cntf = (float)cnts[c];
    if (cntf < 1.f) cntf = 1.f;
    float mean = stats[f] / cntf;
    float var  = stats[64 + f] / cntf - mean * mean;
    if (var < 0.f) var = 0.f;
    float rs = rsqrtf(var + BN_EPS);
    float sc = g1[(t * NC + c) * H + f] * rs;
    stats[128 + f] = sc;
    stats[192 + f] = be1[(t * NC + c) * H + f] - mean * sc;
  }
}

// x[node] = relu(bn(h_mid)) @ W2[t,c] + b2[t,c]  (scatter to masked rows only)
__global__ __launch_bounds__(256)
void k_gemm2(int t, int c, const int* __restrict__ cnts, const int* __restrict__ node_list,
             const float* __restrict__ h_mid, const float* __restrict__ stats,
             const float* __restrict__ W2, const float* __restrict__ b2,
             float* __restrict__ x_work) {
  __shared__ _Float16 sA[128 * 72];
  __shared__ _Float16 sW[64 * 72];
  int cnt = cnts[c], coff = cnts[8 + c];
  int rowBase = blockIdx.x * 128;
  if (rowBase >= cnt) return;
  bool fullTile = (rowBase + 128 <= cnt);

  const float* W = W2 + (t * NC + c) * H * H;
  for (int idx = threadIdx.x; idx < H * H; idx += 256) {
    int k = idx >> 6, n = idx & 63;
    sW[n * 72 + k] = (_Float16)W[idx];
  }
  {
    int r  = threadIdx.x >> 1;
    int k0 = (threadIdx.x & 1) * 32;
    int grow = rowBase + r;
    _Float16* dstp = &sA[r * 72 + k0];
    if (grow < cnt) {
      const float* ph = h_mid + grow * H + k0;
      #pragma unroll
      for (int i = 0; i < 32; i++) {
        float val = ph[i] * stats[128 + k0 + i] + stats[192 + k0 + i];
        dstp[i] = (_Float16)(val > 0.f ? val : 0.f);
      }
    } else {
      #pragma unroll
      for (int i = 0; i < 32; i++) dstp[i] = (_Float16)0.f;
    }
  }
  __syncthreads();

  int lane = threadIdx.x & 31;
  int wave = threadIdx.x >> 5;
  int m = lane & 15, half = lane >> 4;
  int r0 = wave * 16;
  v8f acc[4] = {};
  #pragma unroll
  for (int kc = 0; kc < 2; kc++) {
    v16h a;
    const _Float16* arow = &sA[(r0 + m) * 72 + kc * 32];
    #pragma unroll
    for (int i = 0; i < 16; i++) a[i] = arow[half * 16 + i];
    #pragma unroll
    for (int nt = 0; nt < 4; nt++) {
      v16h b;
      const _Float16* bcol = &sW[(nt * 16 + m) * 72 + kc * 32 + half * 16];
      #pragma unroll
      for (int i = 0; i < 16; i++) b[i] = bcol[i];
      acc[nt] = __builtin_amdgcn_wmma_f32_16x16x32_f16(false, a, false, b, (short)0,
                                                       acc[nt], false, false);
    }
  }
  const float* bias = b2 + (t * NC + c) * H;
  if (fullTile) {
    int nd[8];
    #pragma unroll
    for (int v = 0; v < 8; v++) nd[v] = node_list[coff + rowBase + r0 + half * 8 + v];
    #pragma unroll
    for (int nt = 0; nt < 4; nt++) {
      int ncol = nt * 16 + m;
      float bv = bias[ncol];
      #pragma unroll
      for (int v = 0; v < 8; v++)
        x_work[nd[v] * H + ncol] = acc[nt][v] + bv;
    }
  } else {
    int nd[8];
    #pragma unroll
    for (int v = 0; v < 8; v++) {
      int grow = rowBase + r0 + half * 8 + v;
      nd[v] = (grow < cnt) ? node_list[coff + grow] : -1;
    }
    #pragma unroll
    for (int nt = 0; nt < 4; nt++) {
      int ncol = nt * 16 + m;
      float bv = bias[ncol];
      #pragma unroll
      for (int v = 0; v < 8; v++)
        if (nd[v] >= 0) x_work[nd[v] * H + ncol] = acc[nt][v] + bv;
    }
  }
}

__global__ void k_pool(int t, const float* __restrict__ x_work,
                       const long long* __restrict__ batch, float* __restrict__ pooled) {
  int i = blockIdx.x * blockDim.x + threadIdx.x;
  if (i < NN * H) {
    int node = i >> 6, f = i & 63;
    int g = (int)batch[node];
    atomicAdd(&pooled[g * (NL * H) + t * H + f], x_work[i]);
  }
}

__global__ __launch_bounds__(256)
void k_final(const float* __restrict__ pooled, const float* __restrict__ Wp1,
             const float* __restrict__ bp1, const float* __restrict__ gp,
             const float* __restrict__ bep, const float* __restrict__ Wp2,
             const float* __restrict__ bp2, float* __restrict__ out) {
  __shared__ float z1[NG * H];
  __shared__ float sc[H], sh[H];
  int tid = threadIdx.x;
  for (int idx = tid; idx < NG * H; idx += 256) {
    int g = idx >> 6, f = idx & 63;
    float acc = bp1[f];
    for (int k = 0; k < NL * H; k++) acc += pooled[g * (NL * H) + k] * Wp1[k * H + f];
    z1[idx] = acc;
  }
  __syncthreads();
  if (tid < H) {
    float s = 0.f, sq = 0.f;
    for (int g = 0; g < NG; g++) { float v = z1[g * H + tid]; s += v; sq += v * v; }
    float mean = s * (1.f / NG);
    float var = sq * (1.f / NG) - mean * mean;
    if (var < 0.f) var = 0.f;
    float rs = rsqrtf(var + BN_EPS);
    sc[tid] = gp[tid] * rs;
    sh[tid] = bep[tid] - mean * sc[tid];
  }
  __syncthreads();
  for (int idx = tid; idx < NG * H; idx += 256) {
    float v = z1[idx] * sc[idx & 63] + sh[idx & 63];
    z1[idx] = v > 0.f ? v : 0.f;
  }
  __syncthreads();
  for (int idx = tid; idx < NG * H; idx += 256) {
    int g = idx >> 6, f = idx & 63;
    float acc = bp2[f];
    for (int k = 0; k < H; k++) acc += z1[g * H + k] * Wp2[k * H + f];
    out[idx] = acc;
  }
}

extern "C" void kernel_launch(void* const* d_in, const int* in_sizes, int n_in,
                              void* d_out, int out_size, void* d_ws, size_t ws_size,
                              hipStream_t stream) {
  (void)in_sizes; (void)n_in; (void)out_size; (void)ws_size;
  const float*     x      = (const float*)d_in[0];
  const int*       labels = (const int*)d_in[1];
  const long long* ei     = (const long long*)d_in[2];
  const long long* batch  = (const long long*)d_in[3];
  const float* W1  = (const float*)d_in[4];
  const float* b1  = (const float*)d_in[5];
  const float* g1  = (const float*)d_in[6];
  const float* be1 = (const float*)d_in[7];
  const float* W2  = (const float*)d_in[8];
  const float* b2  = (const float*)d_in[9];
  const float* Wp1 = (const float*)d_in[10];
  const float* bp1 = (const float*)d_in[11];
  const float* gp  = (const float*)d_in[12];
  const float* bep = (const float*)d_in[13];
  const float* Wp2 = (const float*)d_in[14];
  const float* bp2 = (const float*)d_in[15];
  float* out = (float*)d_out;

  char* ws = (char*)d_ws;
  float* x_work = (float*)ws;           ws += (size_t)NN * H * sizeof(float);
  float* agg    = (float*)ws;           ws += (size_t)NN * H * sizeof(float);
  float* h_mid  = (float*)ws;           ws += (size_t)NN * H * sizeof(float);
  float* pooled = (float*)ws;           ws += (size_t)NG * NL * H * sizeof(float);
  float* stats  = (float*)ws;           ws += 256 * sizeof(float);
  int* node_list = (int*)ws;            ws += (size_t)NN * sizeof(int);
  int* edge_list = (int*)ws;            ws += (size_t)NE * sizeof(int);
  int* cnts      = (int*)ws;            ws += 32 * sizeof(int);

  const int T = 256;
  int nb_init = (NN * H + T - 1) / T;
  int nb_ne   = (NN + NE + T - 1) / T;
  int nb_gemm = (NN + 127) / 128;   // worst case; blocks past device count early-exit

  k_init<<<nb_init, T, 0, stream>>>(x, x_work, pooled, cnts);
  k_bucket_count<<<nb_ne, T, 0, stream>>>(labels, ei, cnts);
  k_bucket_offsets<<<1, 32, 0, stream>>>(cnts);
  k_bucket_fill<<<nb_ne, T, 0, stream>>>(labels, ei, cnts, node_list, edge_list);

  for (int t = 0; t < NL; t++) {
    for (int c = 0; c < NC; c++) {
      k_zero_agg<<<2048, T, 0, stream>>>(c, cnts, node_list, agg, stats);
      k_aggregate<<<4096, T, 0, stream>>>(c, cnts, edge_list, ei, x_work, agg);
      k_gemm1<<<nb_gemm, T, 0, stream>>>(t, c, cnts, node_list, agg, x_work, W1, b1, h_mid, stats);
      k_bn_finalize<<<1, 64, 0, stream>>>(t, c, cnts, g1, be1, stats);
      k_gemm2<<<nb_gemm, T, 0, stream>>>(t, c, cnts, node_list, h_mid, stats, W2, b2, x_work);
    }
    k_pool<<<nb_init, T, 0, stream>>>(t, x_work, batch, pooled);
  }
  k_final<<<1, T, 0, stream>>>(pooled, Wp1, bp1, gp, bep, Wp2, bp2, out);
}
